// PointTTTBlock_8761733284539
// MI455X (gfx1250) — matrix operand order repaired and checked by reading, not compiled, not run on hardware
//
#include <hip/hip_runtime.h>
#include <hip/hip_bf16.h>

typedef __attribute__((ext_vector_type(16))) __bf16 bf16x16;
typedef __attribute__((ext_vector_type(8)))  float  f32x8;
typedef __attribute__((ext_vector_type(4)))  unsigned int u32x4;

constexpr int C_DIM = 384;
constexpr int C_NH  = 6;
constexpr int C_HD  = 64;
constexpr int C_K   = 64;
constexpr int C_MAT = C_DIM * C_DIM;          // 147456
constexpr int LDX   = 392;                    // padded LDS row stride (bank-conflict free, 16B aligned)

// ---------------- workspace (bf16) layout, offsets in elements ----------------
__host__ __device__ constexpr size_t OFF_QT (int d) { return (size_t)d * 4 * C_MAT + 0 * (size_t)C_MAT; }
__host__ __device__ constexpr size_t OFF_VT (int d) { return (size_t)d * 4 * C_MAT + 1 * (size_t)C_MAT; }
__host__ __device__ constexpr size_t OFF_GT (int d) { return (size_t)d * 4 * C_MAT + 2 * (size_t)C_MAT; }
__host__ __device__ constexpr size_t OFF_WOT(int d) { return (size_t)d * 4 * C_MAT + 3 * (size_t)C_MAT; }
__host__ __device__ constexpr size_t OFF_W1T(int d) { return (size_t)8 * C_MAT + (size_t)d * C_NH * C_HD * C_HD; }
__host__ __device__ constexpr size_t OFF_FC1()      { return (size_t)8 * C_MAT + 2ull * C_NH * C_HD * C_HD; }
__host__ __device__ constexpr size_t OFF_FC2()      { return OFF_FC1() + C_MAT; }
__host__ __device__ constexpr size_t OFF_GW ()      { return OFF_FC2() + C_MAT; }

struct TTTW {
  const float *q, *v, *cqw, *cqb, *ckw, *ckb, *lrw, *lrb, *lti, *lnw, *lnb,
              *W1, *b1, *pw, *pb, *g, *wo;
};
struct FEPW {
  const float *fc1w, *fc1b, *fc2w, *fc2b, *gw, *gb, *dw1w, *dw1b, *dw2w, *dw2b;
};
struct Args {
  const float* data;
  TTTW t[2];
  FEPW f;
  const float *gate_f, *gate_b;
  const __bf16* ws;
  float* out;
};

// ---------------- WMMA helpers ----------------
__device__ inline bf16x16 load_frag_a(const __bf16* base, int ld) {
  int lane = threadIdx.x & 31;
  const __bf16* p = base + (lane & 15) * ld + ((lane >> 4) << 3);
  union { bf16x16 v; u32x4 q[2]; } u;
  u.q[0] = *(const u32x4*)p;
  u.q[1] = *(const u32x4*)(p + 16);
  return u.v;
}
__device__ inline bf16x16 load_frag_b(const __bf16* base, int ld) {
  int lane = threadIdx.x & 31;
  const __bf16* p = base + (lane & 15) * ld + ((lane >> 4) << 3);
  union { bf16x16 v; u32x4 q[2]; } u;
  u.q[0] = *(const u32x4*)p;
  u.q[1] = *(const u32x4*)(p + 16);
  return u.v;
}
__device__ inline f32x8 wmma_bf(bf16x16 a, bf16x16 b, f32x8 c) {
  return __builtin_amdgcn_wmma_f32_16x16x32_bf16(false, a, false, b, (short)0, c,
                                                 false, false);
}
__device__ inline f32x8 zero8() {
  f32x8 z = {0.f, 0.f, 0.f, 0.f, 0.f, 0.f, 0.f, 0.f};
  return z;
}
__device__ inline int erow() { return ((threadIdx.x & 31) >> 4) << 3; }
__device__ inline int ecol() { return threadIdx.x & 15; }

// 64x64 GEMM, operands (mostly) in LDS: 16 tiles over 8 waves, 2 tiles each
template <typename Epi>
__device__ inline void gemm_mn(const __bf16* A, int lda, const __bf16* B, int ldb,
                               int ntiles_n, int ksteps, Epi&& epi) {
  int wave = threadIdx.x >> 5;
  int total = 4 * ntiles_n;
  for (int t = wave; t < total; t += 8) {
    int mt = (t & 3) << 4;
    int nt = (t >> 2) << 4;
    f32x8 acc = zero8();
    for (int ks = 0; ks < ksteps; ++ks) {
      int k0 = ks << 5;
      acc = wmma_bf(load_frag_a(A + mt * lda + k0, lda),
                    load_frag_b(B + nt * ldb + k0, ldb), acc);
    }
    epi(mt, nt, acc);
  }
}

// 64x384 GEMM, B streamed from global: panel-per-wave, B fragment reused across 4 M-tiles
template <typename Epi>
__device__ inline void gemm_p24(const __bf16* A, int lda, const __bf16* B, int ldb,
                                int ksteps, Epi&& epi) {
  int wave = threadIdx.x >> 5;
  for (int np = wave; np < 24; np += 8) {
    int nt = np << 4;
    f32x8 acc[4] = {zero8(), zero8(), zero8(), zero8()};
    for (int ks = 0; ks < ksteps; ++ks) {
      int k0 = ks << 5;
      bf16x16 b = load_frag_b(B + nt * ldb + k0, ldb);
#pragma unroll
      for (int m = 0; m < 4; ++m)
        acc[m] = wmma_bf(load_frag_a(A + (m << 4) * lda + k0, lda), b, acc[m]);
    }
    for (int m = 0; m < 4; ++m) epi(m << 4, nt, acc[m]);
  }
}

// 64x64 GEMM with K=384, B streamed from global: 4 panels x 2 half-waves, 2 M-tiles each
template <typename Epi>
__device__ inline void gemm_p4h(const __bf16* A, int lda, const __bf16* B, int ldb,
                                int ksteps, Epi&& epi) {
  int wave = threadIdx.x >> 5;
  int nt = (wave >> 1) << 4;
  int mh = (wave & 1) << 1;
  f32x8 acc[2] = {zero8(), zero8()};
  for (int ks = 0; ks < ksteps; ++ks) {
    int k0 = ks << 5;
    bf16x16 b = load_frag_b(B + nt * ldb + k0, ldb);
#pragma unroll
    for (int m = 0; m < 2; ++m)
      acc[m] = wmma_bf(load_frag_a(A + ((mh + m) << 4) * lda + k0, lda), b, acc[m]);
  }
  for (int m = 0; m < 2; ++m) epi((mh + m) << 4, nt, acc[m]);
}

__device__ inline float sigm(float x) { return 1.f / (1.f + __expf(-x)); }
__device__ inline float gelu_t(float x) {
  float x3 = x * x * x;
  return 0.5f * x * (1.f + tanhf(0.7978845608f * (x + 0.044715f * x3)));
}

// ---------------- weight prep: f32 -> bf16 (+transpose where needed) ----------------
__global__ __launch_bounds__(256) void prep_kernel(Args a, __bf16* ws) {
  int i = blockIdx.x * 256 + threadIdx.x;
  if (i < C_MAT) {
    int o = i / C_DIM, c = i - o * C_DIM;
    for (int d = 0; d < 2; ++d) {
      ws[OFF_QT(d)  + i] = (__bf16)a.t[d].q [c * C_DIM + o];
      ws[OFF_VT(d)  + i] = (__bf16)a.t[d].v [c * C_DIM + o];
      ws[OFF_GT(d)  + i] = (__bf16)a.t[d].g [c * C_DIM + o];
      ws[OFF_WOT(d) + i] = (__bf16)a.t[d].wo[c * C_DIM + o];
    }
    ws[OFF_FC1() + i] = (__bf16)a.f.fc1w[i];
    ws[OFF_FC2() + i] = (__bf16)a.f.fc2w[i];
    ws[OFF_GW()  + i] = (__bf16)a.f.gw[i];
  }
  if (i < 2 * C_NH * C_HD * C_HD) {
    int d  = i / (C_NH * C_HD * C_HD);
    int r  = i % (C_NH * C_HD * C_HD);
    int h  = r / (C_HD * C_HD);
    int e  = (r / C_HD) % C_HD;
    int dd = r % C_HD;
    ws[(size_t)8 * C_MAT + i] = (__bf16)a.t[d].W1[h * C_HD * C_HD + dd * C_HD + e];
  }
}

// ---------------- main fused kernel: one workgroup per 64-token block ----------------
__global__ __launch_bounds__(256) void ttt_block(Args a) {
  __shared__ __align__(16) __bf16 XB[C_K * LDX];       // x in pass order (A operand)
  __shared__ __align__(16) __bf16 OUTB[C_K * LDX];     // head outputs / staging
  __shared__ __align__(16) __bf16 HRE[4 * C_K * 72 + 3 * C_K * 64];  // head bufs / FEP G
  __shared__ float lrv[C_NH * C_K];
  __shared__ float tokidx[C_K];

  __bf16* HQ = HRE;                 // ld 72 (roped XQ, A operand)
  __bf16* HK = HRE + 1 * C_K * 72;  // ld 72 (roped XK, A + B operand)
  __bf16* HM = HRE + 2 * C_K * 72;  // ld 72 (-eta*(Attn+1) tril, A operand)
  __bf16* HG = HRE + 3 * C_K * 72;  // ld 72 (grad^T, B operand)
  __bf16* HA = HRE + 4 * C_K * 72;                 // ld 64 (xq_shared slice)
  __bf16* HV = HRE + 4 * C_K * 72 + 1 * C_K * 64;  // ld 64
  __bf16* HZ = HRE + 4 * C_K * 72 + 2 * C_K * 64;  // ld 64 (Z1 / Z1_bar)

  int bid = blockIdx.x;
  int tid = threadIdx.x;
  const float* xg = a.data + (size_t)bid * C_K * C_DIM;
  float* fused = a.out + (size_t)bid * C_K * C_DIM;   // stage fused sum in d_out

  float gsc0 = tanhf(a.gate_f[0]);
  float gsc1 = tanhf(a.gate_b[0]);

  for (int dir = 0; dir < 2; ++dir) {
    const TTTW& P = a.t[dir];
    const __bf16* qT  = a.ws + OFF_QT(dir);
    const __bf16* vT  = a.ws + OFF_VT(dir);
    const __bf16* gT  = a.ws + OFF_GT(dir);
    const __bf16* woT = a.ws + OFF_WOT(dir);
    const __bf16* W1T = a.ws + OFF_W1T(dir);

    for (int i = tid; i < C_K * C_DIM; i += 256) {
      int r = i / C_DIM, c = i - r * C_DIM;
      int sr = dir ? (63 - r) : r;
      XB[r * LDX + c] = (__bf16)xg[sr * C_DIM + c];
    }
    __syncthreads();

    // per-token learning rates + token idx
    for (int j = tid; j < C_NH * C_K; j += 256) {
      int h = j >> 6, t = j & 63;
      float s = P.lrb[h];
      for (int c = 0; c < C_DIM; ++c) s += (float)XB[t * LDX + c] * P.lrw[h * C_DIM + c];
      lrv[j] = sigm(s) * (1.0f / 64.f);    // BASE_LR / HD
    }
    for (int t = tid; t < C_K; t += 256)
      tokidx[t] = fmaxf(1.f / (float)(t + 1) + P.lti[t], 0.f);
    __syncthreads();

    for (int h = 0; h < C_NH; ++h) {
      const float* lnw = P.lnw + h * C_HD;
      const float* lnb = P.lnb + h * C_HD;
      const float* b1h = P.b1  + h * C_HD;

      // xq_shared slice and XV slice for this head
      gemm_p4h(XB, LDX, qT + (size_t)h * C_HD * C_DIM, C_DIM, 12,
               [&](int mt, int nt, const f32x8& acc) {
                 int m0 = mt + erow(), n = nt + ecol();
                 for (int r = 0; r < 8; ++r) HA[(m0 + r) * 64 + n] = (__bf16)acc[r];
               });
      gemm_p4h(XB, LDX, vT + (size_t)h * C_HD * C_DIM, C_DIM, 12,
               [&](int mt, int nt, const f32x8& acc) {
                 int m0 = mt + erow(), n = nt + ecol();
                 for (int r = 0; r < 8; ++r) HV[(m0 + r) * 64 + n] = (__bf16)acc[r];
               });
      __syncthreads();

      // causal depthwise conv (k=4) -> HQ, HK
      for (int i = tid; i < C_K * C_HD; i += 256) {
        int t = i >> 6, c = i & 63, C = h * C_HD + c;
        float aq = P.cqb[C], ak = P.ckb[C];
        for (int k = 0; k < 4; ++k) {
          int tt = t - 3 + k;
          if (tt >= 0) {
            float v = (float)HA[tt * 64 + c];
            aq += P.cqw[C * 4 + k] * v;
            ak += P.ckw[C * 4 + k] * v;
          }
        }
        HQ[t * 72 + c] = (__bf16)aq;
        HK[t * 72 + c] = (__bf16)ak;
      }
      __syncthreads();

      // interleaved RoPE in place
      for (int i = tid; i < C_K * 32; i += 256) {
        int t = i >> 5, aa = i & 31;
        float ang = (float)t * __expf(-((float)aa / 32.f) * 9.2103404f);
        float cs = __cosf(ang), sn = __sinf(ang);
        {
          float e = (float)HQ[t * 72 + 2 * aa], o = (float)HQ[t * 72 + 2 * aa + 1];
          HQ[t * 72 + 2 * aa]     = (__bf16)(e * cs - o * sn);
          HQ[t * 72 + 2 * aa + 1] = (__bf16)(o * cs + e * sn);
        }
        {
          float e = (float)HK[t * 72 + 2 * aa], o = (float)HK[t * 72 + 2 * aa + 1];
          HK[t * 72 + 2 * aa]     = (__bf16)(e * cs - o * sn);
          HK[t * 72 + 2 * aa + 1] = (__bf16)(o * cs + e * sn);
        }
      }
      __syncthreads();

      // Z1 = XK @ W1 + b1
      gemm_mn(HK, 72, W1T + (size_t)h * C_HD * C_HD, C_HD, 4, 2,
              [&](int mt, int nt, const f32x8& acc) {
                int m0 = mt + erow(), n = nt + ecol();
                for (int r = 0; r < 8; ++r)
                  HZ[(m0 + r) * 64 + n] = (__bf16)(acc[r] + b1h[n]);
              });
      __syncthreads();

      // Attn = XQ @ XK^T ; fold eta and tril mask: HM = -(eta*(Attn+1)) on tril
      gemm_mn(HQ, 72, HK, 72, 4, 2, [&](int mt, int nt, const f32x8& acc) {
        int m0 = mt + erow(), n = nt + ecol();
        for (int r = 0; r < 8; ++r) {
          int i2 = m0 + r, j2 = n;
          float eta = tokidx[i2] * lrv[h * 64 + j2];
          float mv = (j2 <= i2) ? eta * (acc[r] + 1.f) : 0.f;
          HM[i2 * 72 + j2] = (__bf16)(-mv);
        }
      });
      __syncthreads();

      // grad = ln_fused_l2_bwd(Z1, XV-XK, ln_w, ln_b); store transposed -> HG[e][t]
      if (tid < C_K) {
        int t = tid;
        float mu = 0.f;
        for (int e = 0; e < C_HD; ++e) mu += (float)HZ[t * 64 + e];
        mu *= (1.f / 64.f);
        float var = 0.f;
        for (int e = 0; e < C_HD; ++e) { float d = (float)HZ[t * 64 + e] - mu; var += d * d; }
        var *= (1.f / 64.f);
        float std_ = sqrtf(var + 1e-6f);
        float inv = 1.f / std_;
        float sg = 0.f, sxg = 0.f;
        for (int e = 0; e < C_HD; ++e) {
          float xh = ((float)HZ[t * 64 + e] - mu) * inv;
          float tgt = (float)HV[t * 64 + e] - (float)HK[t * 72 + e];
          float go = (lnw[e] * xh + lnb[e] - tgt) * lnw[e];
          sg += go; sxg += xh * go;
        }
        float sc = 1.f / (64.f * std_);
        for (int e = 0; e < C_HD; ++e) {
          float xh = ((float)HZ[t * 64 + e] - mu) * inv;
          float tgt = (float)HV[t * 64 + e] - (float)HK[t * 72 + e];
          float go = (lnw[e] * xh + lnb[e] - tgt) * lnw[e];
          HG[e * 72 + t] = (__bf16)((64.f * go - sg - xh * sxg) * sc);
        }
      }
      __syncthreads();

      // Z1_bar = XQ@W1 + b1 + (-M)@grad  (chained WMMA accumulation)
      {
        int wave = tid >> 5;
        for (int t2 = wave; t2 < 16; t2 += 8) {
          int mt = (t2 & 3) << 4, nt = (t2 >> 2) << 4;
          f32x8 acc = zero8();
          for (int ks = 0; ks < 2; ++ks) {
            int k0 = ks << 5;
            acc = wmma_bf(load_frag_a(HQ + mt * 72 + k0, 72),
                          load_frag_b(W1T + (size_t)h * C_HD * C_HD + nt * C_HD + k0, C_HD), acc);
          }
          for (int ks = 0; ks < 2; ++ks) {
            int k0 = ks << 5;
            acc = wmma_bf(load_frag_a(HM + mt * 72 + k0, 72),
                          load_frag_b(HG + nt * 72 + k0, 72), acc);
          }
          int m0 = mt + erow(), n = nt + ecol();
          for (int r = 0; r < 8; ++r)
            HZ[(m0 + r) * 64 + n] = (__bf16)(acc[r] + b1h[n]);
        }
      }
      __syncthreads();

      // out_h = XQ + LN(Z1_bar)
      if (tid < C_K) {
        int t = tid;
        float mu = 0.f, var = 0.f;
        for (int e = 0; e < C_HD; ++e) mu += (float)HZ[t * 64 + e];
        mu *= (1.f / 64.f);
        for (int e = 0; e < C_HD; ++e) { float d = (float)HZ[t * 64 + e] - mu; var += d * d; }
        var *= (1.f / 64.f);
        float inv = 1.f / sqrtf(var + 1e-6f);
        for (int e = 0; e < C_HD; ++e) {
          float ln = lnw[e] * (((float)HZ[t * 64 + e]) - mu) * inv + lnb[e];
          OUTB[t * LDX + h * C_HD + e] = (__bf16)((float)HQ[t * 72 + e] + ln);
        }
      }
      __syncthreads();
    }  // heads

    // post-LN over DIM
    if (tid < C_K) {
      int t = tid;
      float mu = 0.f, var = 0.f;
      for (int c = 0; c < C_DIM; ++c) mu += (float)OUTB[t * LDX + c];
      mu *= (1.f / 384.f);
      for (int c = 0; c < C_DIM; ++c) { float d = (float)OUTB[t * LDX + c] - mu; var += d * d; }
      var *= (1.f / 384.f);
      float inv = 1.f / sqrtf(var + 1e-6f);
      for (int c = 0; c < C_DIM; ++c) {
        float v = P.pw[c] * (((float)OUTB[t * LDX + c]) - mu) * inv + P.pb[c];
        OUTB[t * LDX + c] = (__bf16)v;
      }
    }
    __syncthreads();

    // gate: OUT *= gelu(x @ g_proj)
    gemm_p24(XB, LDX, gT, C_DIM, 12, [&](int mt, int nt, const f32x8& acc) {
      int m0 = mt + erow(), n = nt + ecol();
      for (int r = 0; r < 8; ++r) {
        int idx = (m0 + r) * LDX + n;
        OUTB[idx] = (__bf16)((float)OUTB[idx] * gelu_t(acc[r]));
      }
    });
    __syncthreads();

    // wo projection, accumulate tanh(gate)*out (+x on first pass) into fused (in d_out)
    float gsc = dir ? gsc1 : gsc0;
    gemm_p24(OUTB, LDX, woT, C_DIM, 12, [&](int mt, int nt, const f32x8& acc) {
      int m0 = mt + erow(), n = nt + ecol();
      for (int r = 0; r < 8; ++r) {
        int m = m0 + r;
        int om = dir ? (63 - m) : m;
        int gi = om * C_DIM + nt + ecol() - ecol() + (nt + ecol() - nt); // placeholder removed below
        (void)gi;
      }
      for (int r = 0; r < 8; ++r) {
        int m = m0 + r;
        int om = dir ? (63 - m) : m;
        int gi = om * C_DIM + n;
        float v = gsc * acc[r];
        if (dir == 0) fused[gi] = xg[gi] + v;
        else          fused[gi] += v;
      }
    });
    asm volatile("s_wait_storecnt 0" ::: "memory");
    __syncthreads();
  }  // dir

  // ---------------- FEP ----------------
  const FEPW& F = a.f;
  const __bf16* FC1 = a.ws + OFF_FC1();
  const __bf16* FC2 = a.ws + OFF_FC2();
  const __bf16* GWm = a.ws + OFF_GW();
  __bf16* G = HRE;  // reuse head region, ld = LDX (64*392 = 25088 <= 30720)

  // h = dwconv3(fused) + b  -> OUTB
  for (int i = tid; i < C_K * C_DIM; i += 256) {
    int t = i / C_DIM, c = i - t * C_DIM;
    float s = F.dw1b[c];
    for (int k = 0; k < 3; ++k) {
      int tt = t - 1 + k;
      if (tt >= 0 && tt < C_K) s += F.dw1w[c * 3 + k] * fused[tt * C_DIM + c];
    }
    OUTB[t * LDX + c] = (__bf16)s;
  }
  __syncthreads();

  // dual panel GEMM: u = silu(h@fc1^T + b1) * (h@fc2^T + b2) -> G
  {
    int wave = tid >> 5;
    for (int np = wave; np < 24; np += 8) {
      int nt = np << 4;
      f32x8 a1[4] = {zero8(), zero8(), zero8(), zero8()};
      f32x8 a2[4] = {zero8(), zero8(), zero8(), zero8()};
      for (int ks = 0; ks < 12; ++ks) {
        int k0 = ks << 5;
        bf16x16 b1 = load_frag_b(FC1 + nt * C_DIM + k0, C_DIM);
        bf16x16 b2 = load_frag_b(FC2 + nt * C_DIM + k0, C_DIM);
#pragma unroll
        for (int m = 0; m < 4; ++m) {
          bf16x16 af = load_frag_a(OUTB + (m << 4) * LDX + k0, LDX);
          a1[m] = wmma_bf(af, b1, a1[m]);
          a2[m] = wmma_bf(af, b2, a2[m]);
        }
      }
      int n = nt + ecol();
      for (int m = 0; m < 4; ++m) {
        int m0 = (m << 4) + erow();
        for (int r = 0; r < 8; ++r) {
          float x1 = a1[m][r] + F.fc1b[n];
          float x2 = a2[m][r] + F.fc2b[n];
          G[(m0 + r) * LDX + n] = (__bf16)(x1 * sigm(x1) * x2);
        }
      }
    }
  }
  __syncthreads();

  // g = u @ g_w^T + g_b -> OUTB
  gemm_p24(G, LDX, GWm, C_DIM, 12, [&](int mt, int nt, const f32x8& acc) {
    int m0 = mt + erow(), n = nt + ecol();
    for (int r = 0; r < 8; ++r)
      OUTB[(m0 + r) * LDX + n] = (__bf16)(acc[r] + F.gb[n]);
  });
  __syncthreads();

  // out = fused + dwconv3(g) + b
  for (int i = tid; i < C_K * C_DIM; i += 256) {
    int t = i / C_DIM, c = i - t * C_DIM;
    float s = F.dw2b[c];
    for (int k = 0; k < 3; ++k) {
      int tt = t - 1 + k;
      if (tt >= 0 && tt < C_K) s += F.dw2w[c * 3 + k] * (float)OUTB[tt * LDX + c];
    }
    fused[i] = fused[i] + s;
  }
}

// ---------------- host launcher ----------------
extern "C" void kernel_launch(void* const* d_in, const int* in_sizes, int n_in,
                              void* d_out, int out_size, void* d_ws, size_t ws_size,
                              hipStream_t stream) {
  (void)in_sizes; (void)n_in; (void)out_size; (void)ws_size;
  auto fp = [&](int i) { return (const float*)d_in[i]; };

  Args a{};
  a.data = fp(0);
  // d_in[1] = depth (unused)
  int idx = 2;
  for (int d = 0; d < 2; ++d) {
    TTTW& w = a.t[d];
    w.q   = fp(idx + 0);  w.v   = fp(idx + 1);
    w.cqw = fp(idx + 2);  w.cqb = fp(idx + 3);
    w.ckw = fp(idx + 4);  w.ckb = fp(idx + 5);
    w.lrw = fp(idx + 6);  w.lrb = fp(idx + 7);
    w.lti = fp(idx + 8);
    w.lnw = fp(idx + 9);  w.lnb = fp(idx + 10);
    w.W1  = fp(idx + 11); w.b1  = fp(idx + 12);
    w.pw  = fp(idx + 13); w.pb  = fp(idx + 14);
    w.g   = fp(idx + 15); w.wo  = fp(idx + 16);
    idx += 17;
  }
  FEPW& f = a.f;
  f.fc1w = fp(idx + 0); f.fc1b = fp(idx + 1);
  f.fc2w = fp(idx + 2); f.fc2b = fp(idx + 3);
  f.gw   = fp(idx + 4); f.gb   = fp(idx + 5);
  f.dw1w = fp(idx + 6); f.dw1b = fp(idx + 7);
  f.dw2w = fp(idx + 8); f.dw2b = fp(idx + 9);
  idx += 10;
  a.gate_f = fp(idx);
  a.gate_b = fp(idx + 1);
  a.ws  = (const __bf16*)d_ws;
  a.out = (float*)d_out;

  __bf16* ws = (__bf16*)d_ws;
  hipLaunchKernelGGL(prep_kernel, dim3(576), dim3(256), 0, stream, a, ws);
  hipLaunchKernelGGL(ttt_block, dim3(512), dim3(256), 0, stream, a);
}
